// Block_40106404610136
// MI455X (gfx1250) — matrix operand order repaired and checked by reading, not compiled, not run on hardware
//
#include <hip/hip_runtime.h>
#include <hip/hip_bf16.h>
#include <cmath>

// ---------------- problem constants ----------------
#define NE   384          // n_embd
#define NH   6            // heads
#define HS   64           // head size
#define BB   64           // batch
#define TT   256          // seq len
#define BT   (BB * TT)    // 16384 tokens
#define FF   (4 * NE)     // 1536

typedef __attribute__((ext_vector_type(16))) _Float16 v16h;
typedef __attribute__((ext_vector_type(8)))  float    v8f;
typedef __attribute__((ext_vector_type(4)))  unsigned uint4v;
typedef __attribute__((ext_vector_type(8)))  unsigned uint8v;

union HV16 { v16h v; _Float16 h[16]; };

static __device__ inline v8f v8f_zero() { v8f z = {}; return z; }

// low 32 bits of a flat pointer into __shared__ == workgroup-relative LDS byte addr
static __device__ inline unsigned lds_off(const void* p) {
    return (unsigned)(uintptr_t)p;
}

// ---------------- Tensor Data Mover: 2D tile (f16) -> LDS, with LDS row padding.
// D# group0: count=1 | lds_addr | global_addr(57b) | type=2
// D# group1: data_size=2B, pad_enable, pad_interval/amount, tensor/tile dims, dim0 stride.
// pad_iv code: interval = 2^(code+1) DWORDs; pad_amt code: amount = code+1 DWORDs.
static __device__ inline void tdm_load_2d(unsigned lds_addr, const void* gptr,
                                          unsigned tile_w, unsigned tile_h,
                                          unsigned row_stride,
                                          unsigned pad_iv, unsigned pad_amt) {
    unsigned long long ga = (unsigned long long)(uintptr_t)gptr;
    uint4v g0;
    g0[0] = 1u;                                   // one valid user descriptor
    g0[1] = lds_addr;                             // LDS byte address
    g0[2] = (unsigned)ga;                         // global addr [31:0]
    g0[3] = (unsigned)(ga >> 32) | (2u << 30);    // global addr [56:32] | type=2 (image)
    uint8v g1;
    g1[0] = (1u << 16)                            // data_size = 2 bytes
          | (1u << 20)                            // pad_enable
          | (pad_iv << 22) | (pad_amt << 25);
    g1[1] = (row_stride & 0xFFFFu) << 16;         // tensor_dim0[15:0]   (bits 63:48)
    g1[2] = (row_stride >> 16);                   // tensor_dim0[31:16]; tensor_dim1[15:0]=0
    g1[3] = 0x10u | ((tile_w & 0xFFFFu) << 16);   // tensor_dim1 = 1<<20 ; tile_dim0
    g1[4] = tile_h & 0xFFFFu;                     // tile_dim1 ; tile_dim2 = 0
    g1[5] = row_stride;                           // tensor_dim0_stride [31:0] (elements)
    g1[6] = 0u;                                   // stride[47:32]=0 ; dim1_stride lo = 0
    g1[7] = 0u;
    asm volatile("tensor_load_to_lds %0, %1" :: "s"(g0), "s"(g1) : "memory");
}

static __device__ inline void wait_lds_idle() {
    asm volatile("s_wait_dscnt 0x0" ::: "memory");
}

// 16-lane butterfly reductions inside each half-wave via ds_swizzle
// (group-of-32 mode: imm = xor<<10 | and_mask 0x1f).
template <int XOR>
static __device__ inline float lane_xor_f(float x) {
    int i = __builtin_amdgcn_ds_swizzle(__float_as_int(x), (XOR << 10) | 0x1f);
    return __int_as_float(i);
}
static __device__ inline float rowmax16(float x) {
    x = fmaxf(x, lane_xor_f<1>(x));
    x = fmaxf(x, lane_xor_f<2>(x));
    x = fmaxf(x, lane_xor_f<4>(x));
    x = fmaxf(x, lane_xor_f<8>(x));
    return x;
}
static __device__ inline float rowsum16(float x) {
    x += lane_xor_f<1>(x);
    x += lane_xor_f<2>(x);
    x += lane_xor_f<4>(x);
    x += lane_xor_f<8>(x);
    return x;
}

// ---------------- weight conversion ----------------
__global__ void cvt_f32_f16(const float* __restrict__ in, _Float16* __restrict__ out, int n) {
    int i = blockIdx.x * blockDim.x + threadIdx.x;
    if (i < n) out[i] = (_Float16)in[i];
}

// (H, C, D) fp32 -> (C, H*D) f16  so QKV become plain GEMMs
__global__ void cvt_qkv_f16(const float* __restrict__ in, _Float16* __restrict__ out) {
    int i = blockIdx.x * blockDim.x + threadIdx.x;
    if (i >= NH * NE * HS) return;
    int d = i % HS;
    int c = (i / HS) % NE;
    int h = i / (HS * NE);
    out[(size_t)c * NE + h * HS + d] = (_Float16)in[i];
}

// ---------------- fused LayerNorm -> f16 ----------------
__global__ __launch_bounds__(128)
void ln_to_f16(const float* __restrict__ x, const float* __restrict__ sc,
               const float* __restrict__ bi, _Float16* __restrict__ out) {
    __shared__ float red[128];
    const int row = blockIdx.x, tid = threadIdx.x;
    const float* xr = x + (size_t)row * NE;

    float s = 0.f;
    for (int i = tid; i < NE; i += 128) s += xr[i];
    red[tid] = s; __syncthreads();
    #pragma unroll
    for (int off = 64; off > 0; off >>= 1) {
        if (tid < off) red[tid] += red[tid + off];
        __syncthreads();
    }
    const float mean = red[0] * (1.f / NE);
    __syncthreads();

    float v = 0.f;
    for (int i = tid; i < NE; i += 128) { float d = xr[i] - mean; v += d * d; }
    red[tid] = v; __syncthreads();
    #pragma unroll
    for (int off = 64; off > 0; off >>= 1) {
        if (tid < off) red[tid] += red[tid + off];
        __syncthreads();
    }
    const float rstd = rsqrtf(red[0] * (1.f / NE) + 1e-6f);

    _Float16* orow = out + (size_t)row * NE;
    for (int i = tid; i < NE; i += 128)
        orow[i] = (_Float16)((xr[i] - mean) * rstd * sc[i] + bi[i]);
}

// ---------------- generic WMMA GEMM with TDM double buffering ----------------
// Out(MxN) = A(MxK) * B(KxN) [+bias] [ReLU] [+res fp32] -> out32 and/or out16
// block = 128 threads (4 wave32), tile 64x64, K-step 32.
// Each wave DMAs its own A slice (16 rows x 32) and B slice (8 rows x 64) via TDM
// into the next buffer while WMMAs consume the current one.
template <bool BIAS, bool RELU, bool RES, bool OUT32, bool OUT16>
__global__ __launch_bounds__(128)
void gemm_wmma(const _Float16* __restrict__ A, int lda,
               const _Float16* __restrict__ Bw, int ldb,
               const float* __restrict__ bias,
               const float* __restrict__ res,
               float* __restrict__ out32,
               _Float16* __restrict__ out16,
               int N, int K) {
    __shared__ _Float16 sA[2][64 * 40];   // rows padded to 40 halves (TDM pad 16+4 DW)
    __shared__ _Float16 sB[2][32 * 72];   // rows padded to 72 halves (TDM pad 32+4 DW)

    const int tid  = threadIdx.x;
    const int lane = tid & 31;
    const int wave = tid >> 5;
    const int m0 = blockIdx.x * 64;
    const int n0 = blockIdx.y * 64;

    v8f acc[4];
    #pragma unroll
    for (int i = 0; i < 4; ++i) acc[i] = v8f_zero();

    auto stage = [&](int k0, int buf) {
        // wave-private A slice: rows [wave*16, wave*16+16) of the block tile
        tdm_load_2d(lds_off(&sA[buf][wave * 16 * 40]),
                    A + (size_t)(m0 + wave * 16) * lda + k0,
                    /*tile_w*/32, /*tile_h*/16, (unsigned)lda, /*pad*/3, 3);
        // B slice: rows [wave*8, wave*8+8) of the 32-row K panel
        tdm_load_2d(lds_off(&sB[buf][wave * 8 * 72]),
                    Bw + (size_t)(k0 + wave * 8) * ldb + n0,
                    /*tile_w*/64, /*tile_h*/8, (unsigned)ldb, /*pad*/4, 3);
    };

    stage(0, 0);
    int buf = 0;
    for (int k0 = 0; k0 < K; k0 += 32, buf ^= 1) {
        __builtin_amdgcn_s_wait_tensorcnt(0);   // this wave's TDMs landed
        __syncthreads();                        // everyone's slices visible
        if (k0 + 32 < K) stage(k0 + 32, buf ^ 1);

        // A fragment (16x32) for this wave's row strip
        v16h afrag;
        {
            HV16 u;
            const int m = lane & 15, kk = (lane >> 4) * 16;
            const _Float16* p = &sA[buf][(wave * 16 + m) * 40 + kk];
            #pragma unroll
            for (int i = 0; i < 16; ++i) u.h[i] = p[i];
            afrag = u.v;
        }
        const int nb = lane & 15, kb = (lane >> 4) * 16;
        #pragma unroll
        for (int nt = 0; nt < 4; ++nt) {
            HV16 u;
            const _Float16* p = &sB[buf][kb * 72 + nt * 16 + nb];
            #pragma unroll
            for (int i = 0; i < 16; ++i) u.h[i] = p[i * 72];
            acc[nt] = __builtin_amdgcn_wmma_f32_16x16x32_f16(
                false, afrag, false, u.v, (short)0, acc[nt], false, false);
        }
        __syncthreads();
    }

    // epilogue
    const int nb = lane & 15;
    const int mbase = m0 + wave * 16 + (lane >> 4) * 8;
    #pragma unroll
    for (int nt = 0; nt < 4; ++nt) {
        const int col = n0 + nt * 16 + nb;
        const float badd = BIAS ? bias[col] : 0.f;
        #pragma unroll
        for (int r = 0; r < 8; ++r) {
            const size_t idx = (size_t)(mbase + r) * N + col;
            float v = acc[nt][r] + badd;
            if (RES)  v += res[idx];
            if (RELU) v = v > 0.f ? v : 0.f;
            if (OUT32) out32[idx] = v;
            if (OUT16) out16[idx] = (_Float16)v;
        }
    }
}

// ---------------- flash-style causal attention ----------------
// grid (TT/16, NH, BB), one wave32 per block.  Q/K/V are f16 (BT x NE),
// head h occupies cols [h*64, h*64+64).  O written f16 (BT x NE).
// V chunk staged by TDM, overlapped with the S = Q*K^T WMMAs + softmax.
__global__ __launch_bounds__(32)
void attention_wmma(const _Float16* __restrict__ Q, const _Float16* __restrict__ K,
                    const _Float16* __restrict__ V, _Float16* __restrict__ O,
                    float scale) {
    const int qt = blockIdx.x, h = blockIdx.y, b = blockIdx.z;
    const int lane = threadIdx.x;
    __shared__ _Float16 sP[16 * 40];   // P tile 16x32 (A-layout staging)
    __shared__ _Float16 sV[32 * 72];   // V chunk 32x64 (TDM pad -> stride 72)

    const int rowbase = b * TT + qt * 16;
    const int colbase = h * HS;
    const int nloc = lane & 15;
    const int mrow = (lane >> 4) * 8;

    // Q fragments: 16x64 as two 16x32 A-frags (kept in registers)
    v16h qf[2];
    {
        const int m = lane & 15, k0 = (lane >> 4) * 16;
        const _Float16* p = Q + (size_t)(rowbase + m) * NE + colbase + k0;
        #pragma unroll
        for (int cs = 0; cs < 2; ++cs) {
            HV16 u;
            #pragma unroll
            for (int i = 0; i < 16; ++i) u.h[i] = p[cs * 32 + i];
            qf[cs] = u.v;
        }
    }

    float m_run[8], l_run[8];
    v8f o[4];
    #pragma unroll
    for (int r = 0; r < 8; ++r) { m_run[r] = -1e30f; l_run[r] = 0.f; }
    #pragma unroll
    for (int nt = 0; nt < 4; ++nt) o[nt] = v8f_zero();

    const int jmax = (qt * 16 + 15) >> 5;   // last 32-wide KV chunk touching diagonal
    for (int j = 0; j <= jmax; ++j) {
        const int kv0 = j * 32;

        // async V-chunk DMA (32 x 64) -> sV; drain LDS reads first (tensor ops are
        // unordered vs DS), then overlap the DMA with S computation + softmax.
        wait_lds_idle();
        tdm_load_2d(lds_off(sV), V + (size_t)(b * TT + kv0) * NE + colbase,
                    /*tile_w*/64, /*tile_h*/32, NE, /*pad*/4, 3);

        // S = Q (16x64) * K^T (64x32)  ->  two 16x16 C frags, K-dim split 32+32
        v8f s0 = v8f_zero(), s1 = v8f_zero();
        {
            const int kk0 = (lane >> 4) * 16;
            const _Float16* kbase = K + (size_t)(b * TT + kv0) * NE + colbase;
            #pragma unroll
            for (int cs = 0; cs < 2; ++cs) {
                HV16 u;
                const _Float16* p = kbase + (size_t)nloc * NE + cs * 32 + kk0;
                #pragma unroll
                for (int i = 0; i < 16; ++i) u.h[i] = p[i];
                s0 = __builtin_amdgcn_wmma_f32_16x16x32_f16(
                    false, qf[cs], false, u.v, (short)0, s0, false, false);
            }
            #pragma unroll
            for (int cs = 0; cs < 2; ++cs) {
                HV16 u;
                const _Float16* p = kbase + (size_t)(16 + nloc) * NE + cs * 32 + kk0;
                #pragma unroll
                for (int i = 0; i < 16; ++i) u.h[i] = p[i];
                s1 = __builtin_amdgcn_wmma_f32_16x16x32_f16(
                    false, qf[cs], false, u.v, (short)0, s1, false, false);
            }
        }

        // causal mask + online softmax in C-fragment layout
        #pragma unroll
        for (int r = 0; r < 8; ++r) {
            const int qi = qt * 16 + mrow + r;
            float v0 = s0[r] * scale;
            float v1 = s1[r] * scale;
            if (kv0 + nloc > qi)      v0 = -1e30f;
            if (kv0 + 16 + nloc > qi) v1 = -1e30f;
            const float mx    = rowmax16(fmaxf(v0, v1));
            const float mnew  = fmaxf(m_run[r], mx);
            const float alpha = __expf(m_run[r] - mnew);
            const float p0 = __expf(v0 - mnew);
            const float p1 = __expf(v1 - mnew);
            l_run[r] = l_run[r] * alpha + rowsum16(p0 + p1);
            m_run[r] = mnew;
            #pragma unroll
            for (int nt = 0; nt < 4; ++nt) o[nt][r] *= alpha;
            sP[(mrow + r) * 40 + nloc]      = (_Float16)p0;
            sP[(mrow + r) * 40 + 16 + nloc] = (_Float16)p1;
        }
        __syncthreads();

        // P fragment from LDS (C-layout -> A-layout round trip)
        v16h pfrag;
        {
            HV16 u;
            const int m = lane & 15, k0 = (lane >> 4) * 16;
            const _Float16* p = sP + m * 40 + k0;
            #pragma unroll
            for (int i = 0; i < 16; ++i) u.h[i] = p[i];
            pfrag = u.v;
        }

        __builtin_amdgcn_s_wait_tensorcnt(0);   // V chunk landed in sV

        // O += P (16x32) * V (32x64)
        {
            const int kk0 = (lane >> 4) * 16;
            #pragma unroll
            for (int nt = 0; nt < 4; ++nt) {
                HV16 u;
                const _Float16* p = sV + kk0 * 72 + nt * 16 + nloc;
                #pragma unroll
                for (int i = 0; i < 16; ++i) u.h[i] = p[i * 72];
                o[nt] = __builtin_amdgcn_wmma_f32_16x16x32_f16(
                    false, pfrag, false, u.v, (short)0, o[nt], false, false);
            }
        }
        __syncthreads();
    }

    // normalize and write out (f16, concat-head layout == row-major BT x NE)
    #pragma unroll
    for (int nt = 0; nt < 4; ++nt) {
        #pragma unroll
        for (int r = 0; r < 8; ++r) {
            const size_t idx = (size_t)(rowbase + mrow + r) * NE + colbase + nt * 16 + nloc;
            O[idx] = (_Float16)(o[nt][r] / l_run[r]);
        }
    }
}

// ---------------- launcher ----------------
extern "C" void kernel_launch(void* const* d_in, const int* in_sizes, int n_in,
                              void* d_out, int out_size, void* d_ws, size_t ws_size,
                              hipStream_t stream) {
    const float* x   = (const float*)d_in[0];
    const float* g1  = (const float*)d_in[1];
    const float* be1 = (const float*)d_in[2];
    const float* Wq  = (const float*)d_in[3];
    const float* Wk  = (const float*)d_in[4];
    const float* Wv  = (const float*)d_in[5];
    const float* Wp  = (const float*)d_in[6];
    const float* bp  = (const float*)d_in[7];
    const float* g2  = (const float*)d_in[8];
    const float* be2 = (const float*)d_in[9];
    const float* W1  = (const float*)d_in[10];
    const float* b1  = (const float*)d_in[11];
    const float* W2  = (const float*)d_in[12];
    const float* b2  = (const float*)d_in[13];
    float* out = (float*)d_out;

    // workspace layout
    char* ws = (char*)d_ws;
    size_t cur = 0;
    auto take = [&](size_t bytes) -> char* {
        char* p = ws + cur;
        cur += (bytes + 255) & ~(size_t)255;
        return p;
    };
    _Float16* q16  = (_Float16*)take((size_t)BT * NE * 2);
    _Float16* k16  = (_Float16*)take((size_t)BT * NE * 2);
    _Float16* v16_ = (_Float16*)take((size_t)BT * NE * 2);
    _Float16* h16  = (_Float16*)take((size_t)BT * NE * 2);
    _Float16* ff1  = (_Float16*)take((size_t)BT * FF * 2);
    float*    x2   = (float*)   take((size_t)BT * NE * 4);
    _Float16* wq16 = (_Float16*)take((size_t)NE * NE * 2);
    _Float16* wk16 = (_Float16*)take((size_t)NE * NE * 2);
    _Float16* wv16 = (_Float16*)take((size_t)NE * NE * 2);
    _Float16* wp16 = (_Float16*)take((size_t)NE * NE * 2);
    _Float16* w116 = (_Float16*)take((size_t)NE * FF * 2);
    _Float16* w216 = (_Float16*)take((size_t)FF * NE * 2);
    // safe reuse: h16 dead after QKV GEMMs, q16 dead after attention
    _Float16* attn16 = h16;
    _Float16* h2_16  = q16;

    // 1) weight conversion
    {
        int nqkv = NH * NE * HS;
        cvt_qkv_f16<<<(nqkv + 255) / 256, 256, 0, stream>>>(Wq, wq16);
        cvt_qkv_f16<<<(nqkv + 255) / 256, 256, 0, stream>>>(Wk, wk16);
        cvt_qkv_f16<<<(nqkv + 255) / 256, 256, 0, stream>>>(Wv, wv16);
        cvt_f32_f16<<<(NE * NE + 255) / 256, 256, 0, stream>>>(Wp, wp16, NE * NE);
        cvt_f32_f16<<<(NE * FF + 255) / 256, 256, 0, stream>>>(W1, w116, NE * FF);
        cvt_f32_f16<<<(FF * NE + 255) / 256, 256, 0, stream>>>(W2, w216, FF * NE);
    }

    // 2) LN1
    ln_to_f16<<<BT, 128, 0, stream>>>(x, g1, be1, h16);

    // 3) QKV projections (no bias, f16 out)
    dim3 gemmGridN(BT / 64, NE / 64);
    gemm_wmma<false, false, false, false, true><<<gemmGridN, 128, 0, stream>>>(
        h16, NE, wq16, NE, nullptr, nullptr, nullptr, q16, NE, NE);
    gemm_wmma<false, false, false, false, true><<<gemmGridN, 128, 0, stream>>>(
        h16, NE, wk16, NE, nullptr, nullptr, nullptr, k16, NE, NE);
    gemm_wmma<false, false, false, false, true><<<gemmGridN, 128, 0, stream>>>(
        h16, NE, wv16, NE, nullptr, nullptr, nullptr, v16_, NE, NE);

    // 4) causal attention (note: reference scales by n_embd^-0.5)
    attention_wmma<<<dim3(TT / 16, NH, BB), 32, 0, stream>>>(
        q16, k16, v16_, attn16, 1.0f / sqrtf((float)NE));

    // 5) output projection + bias + residual(x) -> x2 (fp32)
    gemm_wmma<true, false, true, true, false><<<gemmGridN, 128, 0, stream>>>(
        attn16, NE, wp16, NE, bp, x, x2, nullptr, NE, NE);

    // 6) LN2
    ln_to_f16<<<BT, 128, 0, stream>>>(x2, g2, be2, h2_16);

    // 7) FFN up: relu(h2 @ W1 + b1) -> f16
    dim3 gemmGridF(BT / 64, FF / 64);
    gemm_wmma<true, true, false, false, true><<<gemmGridF, 128, 0, stream>>>(
        h2_16, NE, w116, FF, b1, nullptr, nullptr, ff1, FF, NE);

    // 8) FFN down + bias + residual(x2) -> d_out (fp32)
    gemm_wmma<true, false, true, true, false><<<gemmGridN, 128, 0, stream>>>(
        ff1, FF, w216, NE, b2, x2, out, nullptr, NE, FF);
}
